// Rwkv_29326036697504
// MI455X (gfx1250) — matrix operand order repaired and checked by reading, not compiled, not run on hardware
//
#include <hip/hip_runtime.h>
#include <math.h>

// ---------------------------------------------------------------------------
// Types for WMMA fragments (gfx1250, wave32)
// ---------------------------------------------------------------------------
typedef __attribute__((ext_vector_type(16))) _Float16 v16h;
typedef __attribute__((ext_vector_type(8)))  _Float16 h8;
typedef __attribute__((ext_vector_type(4)))  _Float16 h4;
typedef __attribute__((ext_vector_type(8)))  float    v8f;

union AFrag { v16h v; h8 h[2]; };   // POD union of ext-vectors only

#define RWKV_T 2048
#define RWKV_D 1024
#define RWKV_F 4096
#define RWKV_V 50257
#define RWKV_NL 6

// ---------------------------------------------------------------------------
// WMMA GEMM:  C[M,N] = A[M,K] * W[N,K]^T  (+ epilogue)
// A, W fp16 row-major (pre-converted). Double-buffered LDS pipeline.
// Block = 256 threads (8 waves), tile 128x128, K-step 32.
// Wave tile 64x32: 4x2 accumulators, 8 WMMA per K-step.
// ---------------------------------------------------------------------------
#define BM 128
#define BN 128
#define BK 32
#define LDAH 40   // BK + 8 halves padding (80B row stride, 16B aligned chunks)

enum { EPI_NONE = 0, EPI_ADD = 1, EPI_RELU_SQ = 2, EPI_GATED_ADD = 3 };

template <int EPI, typename OutT>
__global__ __launch_bounds__(256) void gemm_f16_wmma_kernel(
    const _Float16* __restrict__ A, const _Float16* __restrict__ W,
    OutT* __restrict__ C, const float* __restrict__ addend,
    const float* __restrict__ gate, int M, int N, int K)
{
    __shared__ __align__(16) _Float16 As[2][BM][LDAH];
    __shared__ __align__(16) _Float16 Bs[2][BN][LDAH];

    const int tid   = threadIdx.x;
    const int lane  = tid & 31;
    const int wave  = tid >> 5;
    const int waveM = (wave >> 2) * 64;   // 0 or 64
    const int waveN = (wave & 3) * 32;    // 0,32,64,96
    const int tileM = blockIdx.y * BM;
    const int tileN = blockIdx.x * BN;

    const int m_lo = lane & 15;
    const int hb   = lane >> 4;           // half-wave select

    v8f acc[4][2];
#pragma unroll
    for (int i = 0; i < 4; ++i)
#pragma unroll
        for (int j = 0; j < 2; ++j)
#pragma unroll
            for (int e = 0; e < 8; ++e) acc[i][j][e] = 0.0f;

    // staging: 256 threads, 2 per row, 16 halves (32B) each
    const int srow = tid >> 1;            // 0..127
    const int scol = (tid & 1) * 16;      // 0 or 16 (halves)

    const _Float16* Ag0 = A + (size_t)(tileM + srow) * K + scol;
    const int gn = tileN + srow;
    const _Float16* Bg0 = W + (size_t)gn * K + scol;
    const bool bvalid = gn < N;

    uint4 ar0, ar1;
    uint4 br0 = make_uint4(0u, 0u, 0u, 0u);
    uint4 br1 = make_uint4(0u, 0u, 0u, 0u);

    // ---- prologue: stage k-step 0 into buffer 0 ----
    {
        const uint4* ap = (const uint4*)Ag0;
        ar0 = ap[0]; ar1 = ap[1];
        if (bvalid) { const uint4* bp = (const uint4*)Bg0; br0 = bp[0]; br1 = bp[1]; }
        *(uint4*)&As[0][srow][scol]     = ar0;
        *(uint4*)&As[0][srow][scol + 8] = ar1;
        *(uint4*)&Bs[0][srow][scol]     = br0;
        *(uint4*)&Bs[0][srow][scol + 8] = br1;
    }
    __syncthreads();

    const int nk = K / BK;
    for (int ks = 0; ks < nk; ++ks) {
        const int  cur  = ks & 1;
        const bool more = (ks + 1) < nk;

        // issue next tile's global loads early (latency hides under WMMA)
        if (more) {
            const size_t off = (size_t)(ks + 1) * BK;
            const uint4* ap = (const uint4*)(Ag0 + off);
            ar0 = ap[0]; ar1 = ap[1];
            if (bvalid) {
                const uint4* bp = (const uint4*)(Bg0 + off);
                br0 = bp[0]; br1 = bp[1];
            }
        }

        // ---- fragments per ISA layouts ----
        AFrag a[4], b[2];
#pragma unroll
        for (int i = 0; i < 4; ++i) {
            const int m = waveM + i * 16 + m_lo;
            a[i].h[0] = *(const h8*)&As[cur][m][8 * hb];        // K = 8*hb + e
            a[i].h[1] = *(const h8*)&As[cur][m][16 + 8 * hb];   // K = 16 + 8*hb + e
        }
#pragma unroll
        for (int j = 0; j < 2; ++j) {
            const int n = waveN + j * 16 + m_lo;
            b[j].h[0] = *(const h8*)&Bs[cur][n][8 * hb];        // K = 8*hb + e
            b[j].h[1] = *(const h8*)&Bs[cur][n][16 + 8 * hb];
        }

#pragma unroll
        for (int i = 0; i < 4; ++i)
#pragma unroll
            for (int j = 0; j < 2; ++j)
                acc[i][j] = __builtin_amdgcn_wmma_f32_16x16x32_f16(
                    false, a[i].v, false, b[j].v, (short)0, acc[i][j], false, false);

        if (more) {
            const int nb = cur ^ 1;
            *(uint4*)&As[nb][srow][scol]     = ar0;
            *(uint4*)&As[nb][srow][scol + 8] = ar1;
            *(uint4*)&Bs[nb][srow][scol]     = br0;
            *(uint4*)&Bs[nb][srow][scol + 8] = br1;
        }
        __syncthreads();
    }

    // ---- epilogue store (C layout: M = e + 8*hb, N = lane&15) ----
#pragma unroll
    for (int i = 0; i < 4; ++i)
#pragma unroll
        for (int j = 0; j < 2; ++j) {
            const int col = tileN + waveN + j * 16 + m_lo;
            if (col >= N) continue;
            const int rbase = tileM + waveM + i * 16 + 8 * hb;
#pragma unroll
            for (int e = 0; e < 8; ++e) {
                const size_t idx = (size_t)(rbase + e) * N + col;
                const float av = acc[i][j][e];
                if (EPI == EPI_NONE) {
                    C[idx] = (OutT)av;
                } else if (EPI == EPI_ADD) {
                    C[idx] = (OutT)(addend[idx] + av);
                } else if (EPI == EPI_RELU_SQ) {
                    const float t = av > 0.f ? av : 0.f;
                    C[idx] = (OutT)(t * t);
                } else {  // EPI_GATED_ADD: out = addend + sigmoid(gate)*acc
                    const float g = gate[idx];
                    C[idx] = (OutT)(addend[idx] + av * (1.f / (1.f + expf(-g))));
                }
            }
        }
}

// ---------------------------------------------------------------------------
// fp32 -> fp16 conversion (vectorized; n multiple of 4)
// ---------------------------------------------------------------------------
__global__ void cvt_f32_f16_kernel(const float* __restrict__ in,
                                   _Float16* __restrict__ out, int n4)
{
    for (int i = blockIdx.x * blockDim.x + threadIdx.x; i < n4;
         i += gridDim.x * blockDim.x) {
        const float4 f = ((const float4*)in)[i];
        h4 hv = { (_Float16)f.x, (_Float16)f.y, (_Float16)f.z, (_Float16)f.w };
        ((h4*)out)[i] = hv;
    }
}

// ---------------------------------------------------------------------------
// Embedding gather
// ---------------------------------------------------------------------------
__global__ void gather_emb_kernel(const int* __restrict__ tokens,
                                  const float* __restrict__ emb,
                                  float* __restrict__ x, int total, int D)
{
    for (int idx = blockIdx.x * blockDim.x + threadIdx.x; idx < total;
         idx += gridDim.x * blockDim.x) {
        const int t = idx / D;
        const int d = idx - t * D;
        x[idx] = emb[(size_t)tokens[t] * D + d];
    }
}

// ---------------------------------------------------------------------------
// LayerNorm over last dim (one block per row); OutT = float or _Float16
// ---------------------------------------------------------------------------
template <typename OutT>
__global__ __launch_bounds__(256) void layernorm_kernel(
    const float* __restrict__ x, const float* __restrict__ g,
    const float* __restrict__ b, OutT* __restrict__ out, int D)
{
    __shared__ float s1[256];
    __shared__ float s2[256];
    const int t = blockIdx.x;
    const float* xr = x + (size_t)t * D;
    OutT* orow = out + (size_t)t * D;

    float sum = 0.f, sq = 0.f;
    for (int i = threadIdx.x; i < D; i += 256) {
        const float v = xr[i];
        sum += v; sq += v * v;
    }
    s1[threadIdx.x] = sum; s2[threadIdx.x] = sq;
    __syncthreads();
    for (int s = 128; s > 0; s >>= 1) {
        if (threadIdx.x < s) {
            s1[threadIdx.x] += s1[threadIdx.x + s];
            s2[threadIdx.x] += s2[threadIdx.x + s];
        }
        __syncthreads();
    }
    const float mu  = s1[0] / D;
    const float var = s2[0] / D - mu * mu;
    const float inv = rsqrtf(var + 1e-5f);
    for (int i = threadIdx.x; i < D; i += 256)
        orow[i] = (OutT)((xr[i] - mu) * inv * g[i] + b[i]);
}

// ---------------------------------------------------------------------------
// time-shift + token mix; fp32 in, fp16 out (GEMM A operands)
// ---------------------------------------------------------------------------
__global__ void mix3_kernel(const float* __restrict__ h,
                            const float* __restrict__ mk,
                            const float* __restrict__ mv,
                            const float* __restrict__ mr,
                            _Float16* __restrict__ xk, _Float16* __restrict__ xv,
                            _Float16* __restrict__ xr, int total, int D)
{
    for (int idx = blockIdx.x * blockDim.x + threadIdx.x; idx < total;
         idx += gridDim.x * blockDim.x) {
        const int t = idx / D;
        const int d = idx - t * D;
        const float hv = h[idx];
        const float lx = (t > 0) ? h[idx - D] : 0.f;
        float a;
        a = mk[d]; xk[idx] = (_Float16)(hv * a + lx * (1.f - a));
        a = mv[d]; xv[idx] = (_Float16)(hv * a + lx * (1.f - a));
        a = mr[d]; xr[idx] = (_Float16)(hv * a + lx * (1.f - a));
    }
}

__global__ void mix2_kernel(const float* __restrict__ h,
                            const float* __restrict__ mk,
                            const float* __restrict__ mr,
                            _Float16* __restrict__ xk, _Float16* __restrict__ xr,
                            int total, int D)
{
    for (int idx = blockIdx.x * blockDim.x + threadIdx.x; idx < total;
         idx += gridDim.x * blockDim.x) {
        const int t = idx / D;
        const int d = idx - t * D;
        const float hv = h[idx];
        const float lx = (t > 0) ? h[idx - D] : 0.f;
        float a;
        a = mk[d]; xk[idx] = (_Float16)(hv * a + lx * (1.f - a));
        a = mr[d]; xr[idx] = (_Float16)(hv * a + lx * (1.f - a));
    }
}

// ---------------------------------------------------------------------------
// WKV linear-attention scan (fp32 state), fused sigmoid(r) gate, fp16 out.
// One thread per channel; loads coalesced; unroll 4 hoists the independent
// loads ahead of the serial recurrence.
// ---------------------------------------------------------------------------
__global__ void wkv_gate_kernel(const float* __restrict__ td,
                                const float* __restrict__ tf,
                                const float* __restrict__ k,
                                const float* __restrict__ v,
                                const float* __restrict__ r,
                                _Float16* __restrict__ gated, int T, int D)
{
    const int d = blockIdx.x * blockDim.x + threadIdx.x;
    if (d >= D) return;
    const float w = expf(-expf(td[d]));
    const float u = tf[d];
    float n = 0.f, den = 0.f;
#pragma unroll 4
    for (int t = 0; t < T; ++t) {
        const size_t idx = (size_t)t * D + d;
        const float kk = k[idx];
        const float vv = v[idx];
        const float rv = r[idx];
        const float ek  = expf(kk);
        const float euk = expf(u + kk);
        const float out = (n + euk * vv) / (den + euk);
        gated[idx] = (_Float16)(out * (1.f / (1.f + expf(-rv))));
        n   = w * n + ek * vv;
        den = w * den + ek;
    }
}

// ---------------------------------------------------------------------------
// In-place softmax over rows of length V
// ---------------------------------------------------------------------------
__global__ __launch_bounds__(256) void softmax_kernel(float* __restrict__ x, int V)
{
    __shared__ float red[256];
    float* row = x + (size_t)blockIdx.x * V;

    float m = -3.4e38f;
    for (int i = threadIdx.x; i < V; i += 256) m = fmaxf(m, row[i]);
    red[threadIdx.x] = m; __syncthreads();
    for (int s = 128; s > 0; s >>= 1) {
        if (threadIdx.x < s)
            red[threadIdx.x] = fmaxf(red[threadIdx.x], red[threadIdx.x + s]);
        __syncthreads();
    }
    m = red[0]; __syncthreads();

    float sum = 0.f;
    for (int i = threadIdx.x; i < V; i += 256) sum += expf(row[i] - m);
    red[threadIdx.x] = sum; __syncthreads();
    for (int s = 128; s > 0; s >>= 1) {
        if (threadIdx.x < s) red[threadIdx.x] += red[threadIdx.x + s];
        __syncthreads();
    }
    const float inv = 1.f / red[0];
    __syncthreads();
    for (int i = threadIdx.x; i < V; i += 256) row[i] = expf(row[i] - m) * inv;
}

// ---------------------------------------------------------------------------
// Host-side helpers
// ---------------------------------------------------------------------------
static void launch_cvt(const float* in, _Float16* out, size_t n, hipStream_t s)
{
    const int n4 = (int)(n / 4);
    int grid = (n4 + 255) / 256;
    if (grid > 16384) grid = 16384;
    cvt_f32_f16_kernel<<<grid, 256, 0, s>>>(in, out, n4);
}

static dim3 gemm_grid(int M, int N)
{
    return dim3((N + BN - 1) / BN, (M + BM - 1) / BM);
}

// ---------------------------------------------------------------------------
// Orchestration
// Input order (setup_inputs dict order, depth-first):
//   [0] tokens, [1] emb, [2] ln0_g, [3] ln0_b,
//   [4 + L*18 + j] layer L tensor j:
//     0 ln1_g, 1 ln1_b, 2 ln2_g, 3 ln2_b, 4 time_decay, 5 time_first,
//     6 mix_k, 7 mix_v, 8 mix_r, 9 Wk, 10 Wv, 11 Wr, 12 Wo,
//     13 f_mix_k, 14 f_mix_r, 15 f_Wk, 16 f_Wr, 17 f_Wv
//   [112] ln_out_g, [113] ln_out_b, [114] head
// ---------------------------------------------------------------------------
extern "C" void kernel_launch(void* const* d_in, const int* in_sizes, int n_in,
                              void* d_out, int out_size, void* d_ws, size_t ws_size,
                              hipStream_t stream)
{
    (void)in_sizes; (void)n_in; (void)out_size; (void)ws_size;
    const int T = RWKV_T, D = RWKV_D, F = RWKV_F, V = RWKV_V;
    const int TD = T * D;
    const size_t DD = (size_t)D * D;
    const size_t FD = (size_t)F * D;
    const size_t VD = (size_t)V * D;

    const int*   tokens = (const int*)d_in[0];
    const float* emb    = (const float*)d_in[1];
    const float* ln0g   = (const float*)d_in[2];
    const float* ln0b   = (const float*)d_in[3];
    const float* lnog   = (const float*)d_in[4 + RWKV_NL * 18];
    const float* lnob   = (const float*)d_in[5 + RWKV_NL * 18];
    const float* headW  = (const float*)d_in[6 + RWKV_NL * 18];

    // ---- workspace layout ----
    // fp32: x0 x1 hbuf kbuf vbuf rbuf rrbuf        (7*TD floats)
    // fp16: xk xv xr gated hhead (5*TD) | kkact (T*F) | weight slot (V*D)
    float* f32b = (float*)d_ws;
    float* xA    = f32b + 0 * (size_t)TD;
    float* xB    = f32b + 1 * (size_t)TD;
    float* hbuf  = f32b + 2 * (size_t)TD;
    float* kbuf  = f32b + 3 * (size_t)TD;
    float* vbuf  = f32b + 4 * (size_t)TD;
    float* rbuf  = f32b + 5 * (size_t)TD;
    float* rrbuf = f32b + 6 * (size_t)TD;

    _Float16* h16 = (_Float16*)(f32b + 7 * (size_t)TD);
    _Float16* xk16    = h16 + 0 * (size_t)TD;
    _Float16* xv16    = h16 + 1 * (size_t)TD;
    _Float16* xr16    = h16 + 2 * (size_t)TD;
    _Float16* gated16 = h16 + 3 * (size_t)TD;
    _Float16* hhead16 = h16 + 4 * (size_t)TD;
    _Float16* kkact16 = h16 + 5 * (size_t)TD;           // T*F halves
    _Float16* wslot   = kkact16 + (size_t)T * F;        // V*D halves (reused)

    // per-layer weight sub-slots (converted fresh each layer; stream-ordered)
    _Float16* Wk16  = wslot + 0 * DD;
    _Float16* Wv16  = wslot + 1 * DD;
    _Float16* Wr16  = wslot + 2 * DD;
    _Float16* Wo16  = wslot + 3 * DD;
    _Float16* fWk16 = wslot + 4 * DD;                   // F*D
    _Float16* fWr16 = wslot + 4 * DD + FD;              // D*D
    _Float16* fWv16 = wslot + 5 * DD + FD;              // D*F

    const int nTD = TD;
    const int gsElem = 8192;

    // x = LN0(emb[tokens])   (layer-0 normalization replaces x, per reference)
    gather_emb_kernel<<<gsElem, 256, 0, stream>>>(tokens, emb, xB, nTD, D);
    layernorm_kernel<float><<<T, 256, 0, stream>>>(xB, ln0g, ln0b, xA, D);

    for (int L = 0; L < RWKV_NL; ++L) {
        void* const* p = d_in + 4 + L * 18;
        const float* ln1g = (const float*)p[0];
        const float* ln1b = (const float*)p[1];
        const float* ln2g = (const float*)p[2];
        const float* ln2b = (const float*)p[3];
        const float* tdec = (const float*)p[4];
        const float* tfir = (const float*)p[5];
        const float* mk   = (const float*)p[6];
        const float* mv   = (const float*)p[7];
        const float* mr   = (const float*)p[8];

        // convert this layer's weights to fp16 into the shared slot
        launch_cvt((const float*)p[9],  Wk16,  DD, stream);
        launch_cvt((const float*)p[10], Wv16,  DD, stream);
        launch_cvt((const float*)p[11], Wr16,  DD, stream);
        launch_cvt((const float*)p[12], Wo16,  DD, stream);
        launch_cvt((const float*)p[15], fWk16, FD, stream);
        launch_cvt((const float*)p[16], fWr16, DD, stream);
        launch_cvt((const float*)p[17], fWv16, FD, stream);

        // --- attention (WKV) block ---
        layernorm_kernel<float><<<T, 256, 0, stream>>>(xA, ln1g, ln1b, hbuf, D);
        mix3_kernel<<<gsElem, 256, 0, stream>>>(hbuf, mk, mv, mr,
                                                xk16, xv16, xr16, nTD, D);
        gemm_f16_wmma_kernel<EPI_NONE, float><<<gemm_grid(T, D), 256, 0, stream>>>(
            xk16, Wk16, kbuf, nullptr, nullptr, T, D, D);
        gemm_f16_wmma_kernel<EPI_NONE, float><<<gemm_grid(T, D), 256, 0, stream>>>(
            xv16, Wv16, vbuf, nullptr, nullptr, T, D, D);
        gemm_f16_wmma_kernel<EPI_NONE, float><<<gemm_grid(T, D), 256, 0, stream>>>(
            xr16, Wr16, rbuf, nullptr, nullptr, T, D, D);
        wkv_gate_kernel<<<(D + 255) / 256, 256, 0, stream>>>(tdec, tfir, kbuf, vbuf,
                                                             rbuf, gated16, T, D);
        // xB = xA + gated @ Wo^T
        gemm_f16_wmma_kernel<EPI_ADD, float><<<gemm_grid(T, D), 256, 0, stream>>>(
            gated16, Wo16, xB, xA, nullptr, T, D, D);

        // --- channel-mix FFN block ---
        layernorm_kernel<float><<<T, 256, 0, stream>>>(xB, ln2g, ln2b, hbuf, D);
        mix2_kernel<<<gsElem, 256, 0, stream>>>(hbuf, (const float*)p[13],
                                                (const float*)p[14],
                                                xk16, xr16, nTD, D);
        // kkact = relu(xk @ fWk^T)^2  (T x F, fp16)
        gemm_f16_wmma_kernel<EPI_RELU_SQ, _Float16><<<gemm_grid(T, F), 256, 0, stream>>>(
            xk16, fWk16, kkact16, nullptr, nullptr, T, F, D);
        // rr = xr @ fWr^T             (T x D, fp32: epilogue gate input)
        gemm_f16_wmma_kernel<EPI_NONE, float><<<gemm_grid(T, D), 256, 0, stream>>>(
            xr16, fWr16, rrbuf, nullptr, nullptr, T, D, D);
        // xA = xB + sigmoid(rr) * (kkact @ fWv^T)
        gemm_f16_wmma_kernel<EPI_GATED_ADD, float><<<gemm_grid(T, D), 256, 0, stream>>>(
            kkact16, fWv16, xA, xB, rrbuf, T, D, F);
    }

    // final LN (fp16 out) + head GEMM straight into d_out + in-place softmax
    layernorm_kernel<_Float16><<<T, 256, 0, stream>>>(xA, lnog, lnob, hhead16, D);
    launch_cvt(headW, wslot, VD, stream);
    gemm_f16_wmma_kernel<EPI_NONE, float><<<gemm_grid(T, V), 256, 0, stream>>>(
        hhead16, wslot, (float*)d_out, nullptr, nullptr, T, V, D);
    softmax_kernel<<<T, 256, 0, stream>>>((float*)d_out, V);
}
